// AttentionBlock_66503273612000
// MI455X (gfx1250) — matrix operand order repaired and checked by reading, not compiled, not run on hardware
//
#include <hip/hip_runtime.h>

// ---------------------------------------------------------------------------
// AttentionBlock for MI455X (gfx1250): f16 WMMA pipeline with async-to-LDS
// staging (ASYNCcnt double buffering) and coalesced staged stores.
// B=32, C=512, H=W=32 -> N=HW=1024 tokens per batch.
// ---------------------------------------------------------------------------

#define BATCH 32
#define CH    512
#define NT    1024          // HW tokens per batch

typedef __attribute__((ext_vector_type(16))) _Float16 v16h;
typedef __attribute__((ext_vector_type(8)))  _Float16 v8h;
typedef __attribute__((ext_vector_type(8)))  float    v8f;

__device__ __forceinline__ v8f vzero8f() {
    v8f z;
#pragma unroll
    for (int i = 0; i < 8; ++i) z[i] = 0.0f;
    return z;
}

__device__ __forceinline__ v8f wmma_f16(v16h a, v16h b, v8f c) {
    // D = A(16x32) * B(32x16) + C(16x16 f32)
    return __builtin_amdgcn_wmma_f32_16x16x32_f16(
        /*neg_a=*/false, a, /*neg_b=*/false, b,
        /*c_mod=*/(short)0, c, /*reuse_a=*/false, /*reuse_b=*/false);
}

// --- gfx1250 async global->LDS copy (ASYNCcnt tracked) ----------------------
// Generic LDS pointers carry the LDS byte offset in their low 32 bits
// (ISA 10.2), so truncation yields the VDST value.
__device__ __forceinline__ void async_ld_b128(void* lds_ptr, const void* gaddr) {
    unsigned int loff = (unsigned int)(unsigned long long)lds_ptr;
    asm volatile("global_load_async_to_lds_b128 %0, %1, off"
                 :: "v"(loff), "v"(gaddr)
                 : "memory");
}
__device__ __forceinline__ void wait_async0() {
    asm volatile("s_wait_asynccnt 0x0" ::: "memory");
}

// A fragment (16x32 f16, MxK) from row-major f16 buffer: row stride ld elems.
// Lanes 0-15: row=lane, K {k0..k0+7, k0+16..k0+23};
// lanes 16-31: row=lane-16, K {k0+8..k0+15, k0+24..k0+31}.
__device__ __forceinline__ v16h frag_a_g(const _Float16* base, int ld, int row0, int k0) {
    const int lane = threadIdx.x & 31;
    const int hf = lane >> 4, r = lane & 15;
    const _Float16* p = base + (size_t)(row0 + r) * ld + k0 + hf * 8;
    v8h lo = *(const v8h*)(p);
    v8h hi = *(const v8h*)(p + 16);
    v16h o;
#pragma unroll
    for (int i = 0; i < 8; ++i) { o[i] = lo[i]; o[i + 8] = hi[i]; }
    return o;
}

// B fragment (32x16 f16, KxN) from a buffer stored [N, K] row-major.
// Lanes 0-15: col=lane, K {k0..k0+7, k0+16..k0+23}; lanes 16-31: K {k0+8..k0+15, k0+24..k0+31}.
__device__ __forceinline__ v16h frag_b_g(const _Float16* base, int ld, int col0, int k0) {
    const int lane = threadIdx.x & 31;
    const int hf = lane >> 4, r = lane & 15;
    const _Float16* p = base + (size_t)(col0 + r) * ld + k0 + hf * 8;
    v8h lo = *(const v8h*)(p);
    v8h hi = *(const v8h*)(p + 16);
    v16h o;
#pragma unroll
    for (int i = 0; i < 8; ++i) { o[i] = lo[i]; o[i + 8] = hi[i]; }
    return o;
}

// Same as frag_b_g but for an f16 LDS buffer (ds_load_b128 x2).
__device__ __forceinline__ v16h frag_b_lds_h(const _Float16* hb, int ld, int col0, int k0) {
    const int lane = threadIdx.x & 31;
    const int hf = lane >> 4, r = lane & 15;
    const _Float16* p = hb + (col0 + r) * ld + k0 + hf * 8;
    v8h lo = *(const v8h*)(p);
    v8h hi = *(const v8h*)(p + 16);
    v16h o;
#pragma unroll
    for (int i = 0; i < 8; ++i) { o[i] = lo[i]; o[i + 8] = hi[i]; }
    return o;
}

// A fragment from an f32 LDS buffer (softmax probabilities), cvt -> f16.
__device__ __forceinline__ v16h frag_a_lds_f32(const float* s, int ld, int row0, int k0) {
    const int lane = threadIdx.x & 31;
    const int hf = lane >> 4, r = lane & 15;
    const float* p = s + (row0 + r) * ld + k0 + hf * 8;
    v16h o;
#pragma unroll
    for (int i = 0; i < 8; ++i) { o[i] = (_Float16)p[i]; o[i + 8] = (_Float16)p[i + 16]; }
    return o;
}

// ---------------------------------------------------------------------------
// Kernel 1a: x [B,C,N] f32 -> xh [B,N,C] f16, LDS-tiled transpose so both
// global reads and global writes are coalesced. Grid: (N/32, C/32, B).
// ---------------------------------------------------------------------------
__global__ __launch_bounds__(256) void cvt_x_kernel(const float* __restrict__ x,
                                                    _Float16* __restrict__ xh) {
    __shared__ float tile[32][33];
    const int b  = blockIdx.z;
    const int c0 = blockIdx.y * 32;
    const int n0 = blockIdx.x * 32;
    const int col = threadIdx.x & 31;
    const int row = threadIdx.x >> 5;   // 0..7

    const float* xb = x + (size_t)b * CH * NT;
#pragma unroll
    for (int it = 0; it < 4; ++it) {
        const int c = row + it * 8;
        tile[c][col] = xb[(size_t)(c0 + c) * NT + n0 + col];
    }
    __syncthreads();
    _Float16* xo = xh + (size_t)b * NT * CH;
#pragma unroll
    for (int it = 0; it < 4; ++it) {
        const int n = row + it * 8;
        xo[(size_t)(n0 + n) * CH + c0 + col] = (_Float16)tile[col][n];
    }
}

// ---------------------------------------------------------------------------
// Kernel 1b: pack Wq|Wk|Wv|Wo into one f16 buffer [4, C, C] row-major.
// ---------------------------------------------------------------------------
__global__ __launch_bounds__(256) void cvt_w_kernel(const float* __restrict__ Wq,
                                                    const float* __restrict__ Wk,
                                                    const float* __restrict__ Wv,
                                                    const float* __restrict__ Wo,
                                                    _Float16* __restrict__ wh) {
    size_t i = (size_t)blockIdx.x * 256 + threadIdx.x;
    if (i >= (size_t)4 * CH * CH) return;
    int s = (int)(i >> 18);
    size_t j = i & ((size_t)CH * CH - 1);
    const float* w = (s == 0) ? Wq : (s == 1) ? Wk : (s == 2) ? Wv : Wo;
    wh[i] = (_Float16)w[j];
}

// ---------------------------------------------------------------------------
// Kernel 2: projections  Y[o,n] = sum_c W[o,c] * x[n,c]  (per batch, 3x).
// - x tile (128 tokens x 32 k) staged in LDS via global_load_async_to_lds,
//   double buffered on ASYNCcnt (dedupes the 8x intra-WG B-fragment reads).
// - outputs staged in LDS then written as coalesced b128 rows.
// q,k token-major [B,N,C]; v channel-major [B,C,N]. Grid: (N/128, C/128, B).
// ---------------------------------------------------------------------------
#define XT_LD 40            // 32 f16 + pad (80B row stride -> spread banks)
#define ST_LD 136           // 128 f16 + pad (272B row stride)

__global__ __launch_bounds__(256) void proj_kernel(const _Float16* __restrict__ xh,
                                                   const _Float16* __restrict__ wh,
                                                   const float* __restrict__ bq,
                                                   const float* __restrict__ bk,
                                                   const float* __restrict__ bv,
                                                   _Float16* __restrict__ q,
                                                   _Float16* __restrict__ k,
                                                   _Float16* __restrict__ v) {
    __shared__ _Float16 xtile[2][128][XT_LD];   // 20.0 KB double buffer
    __shared__ _Float16 stage[128][ST_LD];      // 34.0 KB output stage

    const int b    = blockIdx.z;
    const int nblk = blockIdx.x * 128;
    const int oblk = blockIdx.y * 128;
    const int tid  = threadIdx.x;
    const int wave = tid >> 5;
    const int lane = tid & 31;
    const int hf = lane >> 4, r = lane & 15;

    const _Float16* xb = xh + (size_t)b * NT * CH;
    const int o0 = oblk + wave * 16;

    for (int p = 0; p < 3; ++p) {
        const _Float16* W = wh + (size_t)p * CH * CH;
        const float* bias = (p == 0) ? bq : (p == 1) ? bk : bv;

        v8f acc[8];
#pragma unroll
        for (int j = 0; j < 8; ++j) acc[j] = vzero8f();

        // Prologue: async-fill buffer 0 with k-slice [0,32).
#pragma unroll
        for (int it = 0; it < 2; ++it) {
            const int slot = tid + it * 256;          // 512 x b128 = 8 KB
            const int n = slot >> 2, ch = slot & 3;   // 4 chunks of 8 f16
            async_ld_b128(&xtile[0][n][ch * 8],
                          xb + (size_t)(nblk + n) * CH + ch * 8);
        }

        for (int ks = 0; ks < CH / 32; ++ks) {
            const int buf = ks & 1;
            wait_async0();          // own async copies for `buf` done
            __syncthreads();        // everyone's copies done / prev buf free

            if (ks + 1 < CH / 32) {
#pragma unroll
                for (int it = 0; it < 2; ++it) {
                    const int slot = tid + it * 256;
                    const int n = slot >> 2, ch = slot & 3;
                    async_ld_b128(&xtile[buf ^ 1][n][ch * 8],
                                  xb + (size_t)(nblk + n) * CH + (ks + 1) * 32 + ch * 8);
                }
            }

            v16h a = frag_a_g(W, CH, o0, ks * 32);
#pragma unroll
            for (int nt = 0; nt < 8; ++nt) {
                v16h bf = frag_b_lds_h(&xtile[buf][0][0], XT_LD, nt * 16, 0);
                acc[nt] = wmma_f16(a, bf, acc[nt]);
            }
        }
        __syncthreads();

        // Stage D tiles into LDS (adds bias), layout chosen per destination.
#pragma unroll
        for (int nt = 0; nt < 8; ++nt) {
#pragma unroll
            for (int g = 0; g < 8; ++g) {
                const int ol = wave * 16 + g + hf * 8;  // local o
                const int nl = nt * 16 + r;             // local n
                const float val = acc[nt][g] + bias[oblk + ol];
                if (p == 2) stage[ol][nl] = (_Float16)val;   // v: [o][n]
                else        stage[nl][ol] = (_Float16)val;   // q,k: [n][o]
            }
        }
        __syncthreads();

        // Coalesced write-out: 128 rows x 128 f16, b128 per thread.
        _Float16* dst;
        size_t rstride;
        size_t rbase;
        if (p == 2) { dst = v; rbase = (size_t)b * CH * NT + (size_t)oblk * NT + nblk; rstride = NT; }
        else        { dst = (p == 0) ? q : k;
                      rbase = (size_t)b * NT * CH + (size_t)nblk * CH + oblk; rstride = CH; }
#pragma unroll
        for (int it = 0; it < 8; ++it) {
            const int t = it * 256 + tid;               // 2048 slots
            const int rr = t >> 4, ch = t & 15;         // 16 x b128 per row
            *(v8h*)(dst + rbase + (size_t)rr * rstride + ch * 8) =
                *(const v8h*)(&stage[rr][ch * 8]);
        }
        __syncthreads();    // stage & xtile[0] free before next projection
    }
}

// ---------------------------------------------------------------------------
// Kernel 3: fused attention + output projection + residual.
// One workgroup = one batch b, one 32-query block. 8 waves.
// LDS: s_buf 32 x S_LD f32 (scores/probs), rowsum[32] f32,
//      h_buf 32 x HB_LD f16 (aliases s_buf, barrier-separated).
// ---------------------------------------------------------------------------
#define S_LD  1032          // 1024 + 8 f32 pad
#define HB_LD 528           // 512 + 16 f16 pad
#define ATTN_LDS_BYTES (32 * S_LD * 4 + 32 * 4)

__global__ __launch_bounds__(256) void attn_kernel(const _Float16* __restrict__ q,
                                                   const _Float16* __restrict__ k,
                                                   const _Float16* __restrict__ v,
                                                   const _Float16* __restrict__ wo,
                                                   const float* __restrict__ bo,
                                                   const float* __restrict__ x,
                                                   float* __restrict__ out) {
    extern __shared__ char smem[];
    float*    s_buf  = (float*)smem;
    float*    rowsum = (float*)(smem + 32 * S_LD * 4);
    _Float16* h_buf  = (_Float16*)smem;                 // aliases s_buf

    const int b  = blockIdx.y;
    const int q0 = blockIdx.x * 32;
    const int wave = threadIdx.x >> 5;
    const int lane = threadIdx.x & 31;
    const int hf = lane >> 4, r = lane & 15;

    const _Float16* qb = q + (size_t)b * NT * CH;       // [N, C]
    const _Float16* kb = k + (size_t)b * NT * CH;       // [N, C]
    const _Float16* vb = v + (size_t)b * CH * NT;       // [C, N]
    const float scale = 0.04419417382415922f;           // 1/sqrt(512)

    // ---- Phase 1: S = scale * Q K^T  (32 x 1024) -> LDS --------------------
    const int rt = wave & 1;
    const int cq = wave >> 1;
    const int qrow0 = q0 + rt * 16;

    for (int g2 = 0; g2 < 2; ++g2) {
        v8f acc[8];
#pragma unroll
        for (int j = 0; j < 8; ++j) acc[j] = vzero8f();

        for (int kk = 0; kk < CH; kk += 32) {
            v16h a = frag_a_g(qb, CH, qrow0, kk);
#pragma unroll
            for (int j = 0; j < 8; ++j) {
                const int m0 = (cq * 16 + g2 * 8 + j) * 16;
                v16h bf = frag_b_g(kb, CH, m0, kk);
                acc[j] = wmma_f16(a, bf, acc[j]);
            }
        }
        const int rowl = rt * 16 + hf * 8;
#pragma unroll
        for (int j = 0; j < 8; ++j) {
            const int m0 = (cq * 16 + g2 * 8 + j) * 16;
#pragma unroll
            for (int g = 0; g < 8; ++g)
                s_buf[(rowl + g) * S_LD + m0 + r] = acc[j][g] * scale;
        }
    }
    __syncthreads();

    // ---- Phase 2: row softmax (unnormalized exp + row sums) ----------------
    for (int rr = 0; rr < 4; ++rr) {
        const int row = wave * 4 + rr;
        float* srow = s_buf + row * S_LD;
        float mx = -3.0e38f;
        for (int j = lane; j < NT; j += 32) mx = fmaxf(mx, srow[j]);
#pragma unroll
        for (int off = 16; off >= 1; off >>= 1) mx = fmaxf(mx, __shfl_xor(mx, off, 32));
        float sum = 0.0f;
        for (int j = lane; j < NT; j += 32) {
            float e = __expf(srow[j] - mx);
            srow[j] = e;
            sum += e;
        }
#pragma unroll
        for (int off = 16; off >= 1; off >>= 1) sum += __shfl_xor(sum, off, 32);
        if (lane == 0) rowsum[row] = sum;
    }
    __syncthreads();

    // ---- Phase 3: H = P V  (32 x 512) --------------------------------------
    const int chq = wave >> 1;
    v8f hacc[8];
#pragma unroll
    for (int j = 0; j < 8; ++j) hacc[j] = vzero8f();

    for (int mm = 0; mm < NT; mm += 32) {
        v16h a = frag_a_lds_f32(s_buf, S_LD, rt * 16, mm);
#pragma unroll
        for (int j = 0; j < 8; ++j) {
            const int c0 = (chq * 8 + j) * 16;
            v16h bf = frag_b_g(vb, NT, c0, mm);
            hacc[j] = wmma_f16(a, bf, hacc[j]);
        }
    }
    float rinv[8];
#pragma unroll
    for (int g = 0; g < 8; ++g) rinv[g] = 1.0f / rowsum[rt * 16 + hf * 8 + g];

    __syncthreads();    // all waves done reading s_buf (P)

    {
        const int rowl = rt * 16 + hf * 8;
#pragma unroll
        for (int j = 0; j < 8; ++j) {
            const int c0 = (chq * 8 + j) * 16;
#pragma unroll
            for (int g = 0; g < 8; ++g)
                h_buf[(rowl + g) * HB_LD + c0 + r] = (_Float16)(hacc[j][g] * rinv[g]);
        }
    }
    __syncthreads();

    // ---- Phase 4: out = x + Wo * h^T + bo  (512 x 32 block) ----------------
    v8f oacc[4][2];
#pragma unroll
    for (int i = 0; i < 4; ++i) { oacc[i][0] = vzero8f(); oacc[i][1] = vzero8f(); }

    for (int kk = 0; kk < CH; kk += 32) {
        v16h bf0 = frag_b_lds_h(h_buf, HB_LD, 0,  kk);
        v16h bf1 = frag_b_lds_h(h_buf, HB_LD, 16, kk);
#pragma unroll
        for (int i = 0; i < 4; ++i) {
            const int o0 = (wave * 4 + i) * 16;
            v16h a = frag_a_g(wo, CH, o0, kk);
            oacc[i][0] = wmma_f16(a, bf0, oacc[i][0]);
            oacc[i][1] = wmma_f16(a, bf1, oacc[i][1]);
        }
    }

    const float* xb2 = x   + (size_t)b * CH * NT;
    float*       ob  = out + (size_t)b * CH * NT;
#pragma unroll
    for (int i = 0; i < 4; ++i) {
        const int o0 = (wave * 4 + i) * 16;
#pragma unroll
        for (int nt = 0; nt < 2; ++nt) {
#pragma unroll
            for (int g = 0; g < 8; ++g) {
                const int o = o0 + g + hf * 8;
                const int n = q0 + nt * 16 + r;
                const size_t idx = (size_t)o * NT + n;
                ob[idx] = xb2[idx] + oacc[i][nt][g] + bo[o];
            }
        }
    }
}

// ---------------------------------------------------------------------------
// Host launcher
// ---------------------------------------------------------------------------
extern "C" void kernel_launch(void* const* d_in, const int* in_sizes, int n_in,
                              void* d_out, int out_size, void* d_ws, size_t ws_size,
                              hipStream_t stream) {
    const float* x  = (const float*)d_in[0];
    const float* Wq = (const float*)d_in[1];
    const float* bq = (const float*)d_in[2];
    const float* Wk = (const float*)d_in[3];
    const float* bk = (const float*)d_in[4];
    const float* Wv = (const float*)d_in[5];
    const float* bv = (const float*)d_in[6];
    const float* Wo = (const float*)d_in[7];
    const float* bo = (const float*)d_in[8];
    float* out = (float*)d_out;

    char* ws = (char*)d_ws;
    // Workspace: xh 32MiB | wh 2MiB | q 32MiB | k 32MiB | v 32MiB
    _Float16* xh = (_Float16*)(ws);
    _Float16* wh = (_Float16*)(ws + 33554432);
    _Float16* qb = (_Float16*)(ws + 35651584);
    _Float16* kb = (_Float16*)(ws + 69206016);
    _Float16* vb = (_Float16*)(ws + 102760448);

    cvt_x_kernel<<<dim3(NT / 32, CH / 32, BATCH), dim3(256), 0, stream>>>(x, xh);
    cvt_w_kernel<<<dim3(4096), dim3(256), 0, stream>>>(Wq, Wk, Wv, Wo, wh);
    proj_kernel<<<dim3(NT / 128, CH / 128, BATCH), dim3(256), 0, stream>>>(
        xh, wh, bq, bk, bv, qb, kb, vb);
    attn_kernel<<<dim3(NT / 32, BATCH), dim3(256), ATTN_LDS_BYTES, stream>>>(
        qb, kb, vb, wh + (size_t)3 * CH * CH, bo, x, out);
}
